// DecoderWithAttention_36197984371410
// MI455X (gfx1250) — matrix operand order repaired and checked by reading, not compiled, not run on hardware
//
#include <hip/hip_runtime.h>
#include <math.h>

// ---------------------------------------------------------------------------
// DecoderWithAttention for MI455X (gfx1250): WMMA f16 GEMMs + fused glue.
// B=64, DEC=12, F=32, L=2, H=512, E=96, T=4
// ---------------------------------------------------------------------------

typedef __attribute__((ext_vector_type(16))) _Float16 v16h;
typedef __attribute__((ext_vector_type(8)))  float    v8f;

#define Bsz 64
#define DECL 12
#define Fin 32
#define Hs  512
#define Es  96
#define Ts  4
#define KX  (Fin + Hs)   // 544
#define G3  (3 * Hs)     // 1536

union FragH { v16h h; unsigned int u[8]; };

__device__ inline float wave_reduce_sum(float v) {
#pragma unroll
  for (int off = 16; off > 0; off >>= 1) v += __shfl_down(v, off, 32);
  return v;
}

__device__ inline float sigmoidf_fast(float x) {
  return 1.0f / (1.0f + __expf(-x));
}

// ---------------------------------------------------------------------------
// Generic WMMA GEMM: C[M x N] (f32) = A[M x K] (f16, row major, lda)
//                                   @ W[N x K]^T (f16, row major, ldw = K)
//                                   + bias[N] (optional, f32)
// block = 128 threads (4 waves); wave w handles M-tile (blockIdx.y*4 + w),
// blockIdx.x handles one 16-wide N tile.  K must be a multiple of 32.
// ---------------------------------------------------------------------------
__global__ void wmma_gemm_f16(float* __restrict__ C,
                              const _Float16* __restrict__ A,
                              const _Float16* __restrict__ W,
                              const float* __restrict__ bias,
                              int M, int N, int K, int lda) {
  const int lane  = threadIdx.x & 31;
  const int wave  = threadIdx.x >> 5;
  const int mbase = (blockIdx.y * 4 + wave) * 16;
  const int nbase = blockIdx.x * 16;
  if (mbase >= M) return;

  // A fragment: lane holds row (mbase + lane%16); K-pair offsets per ISA
  // 16-bit A 16x32 layout (lanes 0-15: K 0..7,16..23 ; lanes 16-31: +8).
  const int koffA = (lane < 16) ? 0 : 8;
  // B fragment: lane holds column (nbase + lane%16); same K layout as A.
  const int koffB = (lane < 16) ? 0 : 8;

  const _Float16* Arow = A + (size_t)(mbase + (lane & 15)) * lda;
  const _Float16* Wrow = W + (size_t)(nbase + (lane & 15)) * K;

  v8f acc = {};
  for (int k0 = 0; k0 < K; k0 += 32) {
    FragH fa, fb;
#pragma unroll
    for (int j = 0; j < 8; ++j) {
      const int kk = koffA + ((j >= 4) ? 16 : 0) + 2 * (j & 3);
      fa.u[j] = *(const unsigned int*)(Arow + k0 + kk);
    }
#pragma unroll
    for (int j = 0; j < 8; ++j) {
      fb.u[j] = *(const unsigned int*)(Wrow + k0 + koffB + ((j >= 4) ? 16 : 0) + 2 * (j & 3));
    }
    if (k0 + 32 < K) {
      // gfx1250 global_prefetch_b8, near-cache scope (pull into WGP$):
      // locality 3 -> scope WGP, so the next K chunk is staged close.
      __builtin_prefetch(Arow + k0 + 32, 0, 3);
      __builtin_prefetch(Wrow + k0 + 32, 0, 3);
    }
    acc = __builtin_amdgcn_wmma_f32_16x16x32_f16(false, fa.h, false, fb.h,
                                                 (short)0, acc, false, false);
  }

  const int n     = nbase + (lane & 15);
  const int mrow0 = mbase + ((lane < 16) ? 0 : 8);
  const float bv  = bias ? bias[n] : 0.0f;
#pragma unroll
  for (int j = 0; j < 8; ++j) {
    C[(size_t)(mrow0 + j) * N + n] = acc[j] + bv;
  }
}

// ---------------------------------------------------------------------------
// Small prep kernels
// ---------------------------------------------------------------------------
__global__ void cvt_f32_to_f16(_Float16* __restrict__ dst,
                               const float* __restrict__ src, int n) {
  int i = blockIdx.x * blockDim.x + threadIdx.x;
  if (i < n) dst[i] = (_Float16)src[i];
}

// dst[r*cols + c] = src[r*src_ld + src_off + c]  (f32 -> f16), for W_attn split
__global__ void cvt_slice_f16(_Float16* __restrict__ dst,
                              const float* __restrict__ src,
                              int rows, int cols, int src_ld, int src_off) {
  int i = blockIdx.x * blockDim.x + threadIdx.x;
  if (i < rows * cols) {
    int r = i / cols, c = i % cols;
    dst[i] = (_Float16)src[(size_t)r * src_ld + src_off + c];
  }
}

__global__ void copy_f32(float* __restrict__ dst,
                         const float* __restrict__ src, int n) {
  int i = blockIdx.x * blockDim.x + threadIdx.x;
  if (i < n) dst[i] = src[i];
}

// cur[b][f] = inputs[b][0][f]
__global__ void init_cur(float* __restrict__ cur,
                         const float* __restrict__ inputs) {
  int i = blockIdx.x * blockDim.x + threadIdx.x;
  if (i < Bsz * Fin) {
    int b = i / Fin, f = i % Fin;
    cur[i] = inputs[(size_t)b * DECL * Fin + f];
  }
}

// ---------------------------------------------------------------------------
// Attention: scores -> softmax -> context (ws); also builds x = [cur, ws] f16
// one block per batch row, 128 threads (4 waves)
// ---------------------------------------------------------------------------
__global__ void attn_kernel(const float* __restrict__ enc_proj,  // B*E*H
                            const float* __restrict__ a,         // B*H
                            const float* __restrict__ v_attn,    // H
                            const float* __restrict__ enc,       // B*E*H (f32)
                            const float* __restrict__ cur,       // B*F
                            float* __restrict__ ws,              // B*H
                            _Float16* __restrict__ xcat) {       // B*KX
  __shared__ float s_w[Es];
  const int b    = blockIdx.x;
  const int lane = threadIdx.x & 31;
  const int wave = threadIdx.x >> 5;
  const float* ep = enc_proj + (size_t)b * Es * Hs;
  const float* ab = a + (size_t)b * Hs;

  for (int e = wave; e < Es; e += 4) {
    const float* row = ep + (size_t)e * Hs;
    float s = 0.0f;
    for (int g = lane; g < Hs; g += 32)
      s += v_attn[g] * tanhf(row[g] + ab[g]);
    s = wave_reduce_sum(s);
    if (lane == 0) s_w[e] = s;
  }
  __syncthreads();
  if (threadIdx.x == 0) {
    float mx = -1e30f;
    for (int e = 0; e < Es; ++e) mx = fmaxf(mx, s_w[e]);
    float sum = 0.0f;
    for (int e = 0; e < Es; ++e) { float ev = __expf(s_w[e] - mx); s_w[e] = ev; sum += ev; }
    float inv = 1.0f / sum;
    for (int e = 0; e < Es; ++e) s_w[e] *= inv;
  }
  __syncthreads();
  const float* eb = enc + (size_t)b * Es * Hs;
  for (int h = threadIdx.x; h < Hs; h += blockDim.x) {
    float s = 0.0f;
    for (int e = 0; e < Es; ++e) s += s_w[e] * eb[(size_t)e * Hs + h];
    ws[(size_t)b * Hs + h] = s;
    xcat[(size_t)b * KX + Fin + h] = (_Float16)s;
  }
  for (int f = threadIdx.x; f < Fin; f += blockDim.x)
    xcat[(size_t)b * KX + f] = (_Float16)cur[(size_t)b * Fin + f];
}

// ---------------------------------------------------------------------------
// Fused GRU gate math: h = (1-z)*n + z*h ; updates f32 state + f16 shadow
// ---------------------------------------------------------------------------
__global__ void gru_gate(const float* __restrict__ gi,
                         const float* __restrict__ gh,
                         float* __restrict__ h_f32,
                         _Float16* __restrict__ h_f16) {
  int i = blockIdx.x * blockDim.x + threadIdx.x;
  if (i >= Bsz * Hs) return;
  int b = i / Hs, j = i % Hs;
  const float* gib = gi + (size_t)b * G3;
  const float* ghb = gh + (size_t)b * G3;
  float r = sigmoidf_fast(gib[j] + ghb[j]);
  float z = sigmoidf_fast(gib[Hs + j] + ghb[Hs + j]);
  float n = tanhf(gib[2 * Hs + j] + r * ghb[2 * Hs + j]);
  float hn = (1.0f - z) * n + z * h_f32[i];
  h_f32[i] = hn;
  h_f16[i] = (_Float16)hn;
}

// ---------------------------------------------------------------------------
// Output projection (T=4 dots of length 2H+F) + autoregressive cur update
// one block per batch row, 128 threads: wave w computes output column w
// ---------------------------------------------------------------------------
__global__ void out_kernel(const float* __restrict__ h1,    // B*H
                           const float* __restrict__ ws,    // B*H
                           float* __restrict__ cur,         // B*F (RW)
                           const float* __restrict__ Wout,  // T x (2H+F)
                           const float* __restrict__ bout,  // T
                           const int* __restrict__ tidx,    // T
                           const float* __restrict__ inputs,
                           float* __restrict__ out,         // B*DEC*T
                           int t) {
  __shared__ float s_out[Ts];
  const int b    = blockIdx.x;
  const int lane = threadIdx.x & 31;
  const int wave = threadIdx.x >> 5;  // == output index tt (4 waves)
  const int D    = 2 * Hs + Fin;      // 1056
  const float* wrow = Wout + (size_t)wave * D;
  float s = 0.0f;
  for (int k = lane; k < D; k += 32) {
    float xv;
    if (k < Hs)            xv = h1[(size_t)b * Hs + k];
    else if (k < 2 * Hs)   xv = ws[(size_t)b * Hs + (k - Hs)];
    else                   xv = cur[(size_t)b * Fin + (k - 2 * Hs)];
    s += xv * wrow[k];
  }
  s = wave_reduce_sum(s);
  if (lane == 0) {
    s += bout[wave];
    s_out[wave] = s;
    out[(size_t)b * DECL * Ts + (size_t)t * Ts + wave] = s;
  }
  __syncthreads();
  // next cur = inputs[:, t, :] with target columns overwritten by out
  for (int f = threadIdx.x; f < Fin; f += blockDim.x)
    cur[(size_t)b * Fin + f] = inputs[(size_t)b * DECL * Fin + (size_t)t * Fin + f];
  __syncthreads();
  if (threadIdx.x < Ts)
    cur[(size_t)b * Fin + tidx[threadIdx.x]] = s_out[threadIdx.x];
}

// ---------------------------------------------------------------------------
// Host launch
// ---------------------------------------------------------------------------
static inline unsigned cdiv(unsigned a, unsigned b) { return (a + b - 1) / b; }

extern "C" void kernel_launch(void* const* d_in, const int* in_sizes, int n_in,
                              void* d_out, int out_size, void* d_ws, size_t ws_size,
                              hipStream_t stream) {
  (void)in_sizes; (void)n_in; (void)out_size; (void)ws_size;
  const float* inputs = (const float*)d_in[0];
  const float* hidden = (const float*)d_in[1];
  const float* enc    = (const float*)d_in[2];
  const int*   tidx   = (const int*)d_in[3];
  const float* W_attn = (const float*)d_in[4];
  const float* b_attn = (const float*)d_in[5];
  const float* v_attn = (const float*)d_in[6];
  const float* Wi0    = (const float*)d_in[7];
  const float* Wh0    = (const float*)d_in[8];
  const float* bi0    = (const float*)d_in[9];
  const float* bh0    = (const float*)d_in[10];
  const float* Wi1    = (const float*)d_in[11];
  const float* Wh1    = (const float*)d_in[12];
  const float* bi1    = (const float*)d_in[13];
  const float* bh1    = (const float*)d_in[14];
  const float* Wout   = (const float*)d_in[15];
  const float* bout   = (const float*)d_in[16];
  float* out = (float*)d_out;

  // ---- carve workspace (256B aligned chunks) ----
  char* base = (char*)d_ws;
  size_t off = 0;
  auto alloc = [&](size_t bytes) -> void* {
    void* p = base + off;
    off += (bytes + 255) & ~(size_t)255;
    return p;
  };
  _Float16* enc_h   = (_Float16*)alloc(sizeof(_Float16) * (size_t)Bsz * Es * Hs);
  _Float16* WaH_h   = (_Float16*)alloc(sizeof(_Float16) * (size_t)Hs * Hs);
  _Float16* WaE_h   = (_Float16*)alloc(sizeof(_Float16) * (size_t)Hs * Hs);
  _Float16* Wi0_h   = (_Float16*)alloc(sizeof(_Float16) * (size_t)G3 * KX);
  _Float16* Wh0_h   = (_Float16*)alloc(sizeof(_Float16) * (size_t)G3 * Hs);
  _Float16* Wi1_h   = (_Float16*)alloc(sizeof(_Float16) * (size_t)G3 * Hs);
  _Float16* Wh1_h   = (_Float16*)alloc(sizeof(_Float16) * (size_t)G3 * Hs);
  float*    encproj = (float*)alloc(sizeof(float) * (size_t)Bsz * Es * Hs);
  float*    a_f32   = (float*)alloc(sizeof(float) * (size_t)Bsz * Hs);
  float*    ws_f32  = (float*)alloc(sizeof(float) * (size_t)Bsz * Hs);
  _Float16* xcat_h  = (_Float16*)alloc(sizeof(_Float16) * (size_t)Bsz * KX);
  float*    h0_f32  = (float*)alloc(sizeof(float) * (size_t)Bsz * Hs);
  float*    h1_f32  = (float*)alloc(sizeof(float) * (size_t)Bsz * Hs);
  _Float16* h0_h    = (_Float16*)alloc(sizeof(_Float16) * (size_t)Bsz * Hs);
  _Float16* h1_h    = (_Float16*)alloc(sizeof(_Float16) * (size_t)Bsz * Hs);
  float*    gi_f32  = (float*)alloc(sizeof(float) * (size_t)Bsz * G3);
  float*    gh_f32  = (float*)alloc(sizeof(float) * (size_t)Bsz * G3);
  float*    cur_f32 = (float*)alloc(sizeof(float) * (size_t)Bsz * Fin);

  const int TPB = 256;
  // ---- one-time conversions ----
  cvt_f32_to_f16<<<cdiv(Bsz * Es * Hs, TPB), TPB, 0, stream>>>(enc_h, enc, Bsz * Es * Hs);
  cvt_f32_to_f16<<<cdiv(G3 * KX, TPB), TPB, 0, stream>>>(Wi0_h, Wi0, G3 * KX);
  cvt_f32_to_f16<<<cdiv(G3 * Hs, TPB), TPB, 0, stream>>>(Wh0_h, Wh0, G3 * Hs);
  cvt_f32_to_f16<<<cdiv(G3 * Hs, TPB), TPB, 0, stream>>>(Wi1_h, Wi1, G3 * Hs);
  cvt_f32_to_f16<<<cdiv(G3 * Hs, TPB), TPB, 0, stream>>>(Wh1_h, Wh1, G3 * Hs);
  cvt_slice_f16<<<cdiv(Hs * Hs, TPB), TPB, 0, stream>>>(WaH_h, W_attn, Hs, Hs, 2 * Hs, 0);
  cvt_slice_f16<<<cdiv(Hs * Hs, TPB), TPB, 0, stream>>>(WaE_h, W_attn, Hs, Hs, 2 * Hs, Hs);
  cvt_f32_to_f16<<<cdiv(Bsz * Hs, TPB), TPB, 0, stream>>>(h0_h, hidden, Bsz * Hs);
  cvt_f32_to_f16<<<cdiv(Bsz * Hs, TPB), TPB, 0, stream>>>(h1_h, hidden + (size_t)Bsz * Hs, Bsz * Hs);
  copy_f32<<<cdiv(Bsz * Hs, TPB), TPB, 0, stream>>>(h0_f32, hidden, Bsz * Hs);
  copy_f32<<<cdiv(Bsz * Hs, TPB), TPB, 0, stream>>>(h1_f32, hidden + (size_t)Bsz * Hs, Bsz * Hs);
  init_cur<<<cdiv(Bsz * Fin, TPB), TPB, 0, stream>>>(cur_f32, inputs);

  // enc_proj = enc @ Wa_e^T  (M = B*E = 6144, N = K = 512)
  wmma_gemm_f16<<<dim3(Hs / 16, (Bsz * Es) / 64), 128, 0, stream>>>(
      encproj, enc_h, WaE_h, nullptr, Bsz * Es, Hs, Hs, Hs);

  // ---- recurrent steps ----
  for (int t = 0; t < DECL; ++t) {
    // a = h1 @ Wa_h^T + b_attn
    wmma_gemm_f16<<<dim3(Hs / 16, 1), 128, 0, stream>>>(
        a_f32, h1_h, WaH_h, b_attn, Bsz, Hs, Hs, Hs);
    // attention scores/softmax/context; builds xcat = [cur, ws] in f16
    attn_kernel<<<Bsz, 128, 0, stream>>>(encproj, a_f32, v_attn, enc, cur_f32,
                                         ws_f32, xcat_h);
    // layer 0: gi = xcat @ Wi0^T + bi0 ; gh = h0 @ Wh0^T + bh0
    wmma_gemm_f16<<<dim3(G3 / 16, 1), 128, 0, stream>>>(
        gi_f32, xcat_h, Wi0_h, bi0, Bsz, G3, KX, KX);
    wmma_gemm_f16<<<dim3(G3 / 16, 1), 128, 0, stream>>>(
        gh_f32, h0_h, Wh0_h, bh0, Bsz, G3, Hs, Hs);
    gru_gate<<<cdiv(Bsz * Hs, TPB), TPB, 0, stream>>>(gi_f32, gh_f32, h0_f32, h0_h);
    // layer 1: gi = h0 @ Wi1^T + bi1 ; gh = h1 @ Wh1^T + bh1
    wmma_gemm_f16<<<dim3(G3 / 16, 1), 128, 0, stream>>>(
        gi_f32, h0_h, Wi1_h, bi1, Bsz, G3, Hs, Hs);
    wmma_gemm_f16<<<dim3(G3 / 16, 1), 128, 0, stream>>>(
        gh_f32, h1_h, Wh1_h, bh1, Bsz, G3, Hs, Hs);
    gru_gate<<<cdiv(Bsz * Hs, TPB), TPB, 0, stream>>>(gi_f32, gh_f32, h1_f32, h1_h);
    // out = [h1, ws, cur] @ Wout^T + bout ; update cur for next step
    out_kernel<<<Bsz, 128, 0, stream>>>(h1_f32, ws_f32, cur_f32, Wout, bout,
                                        tidx, inputs, out, t);
  }
}
